// EGATConv_17815524343864
// MI455X (gfx1250) — matrix operand order repaired and checked by reading, not compiled, not run on hardware
//
#include <hip/hip_runtime.h>

#define IN_FEATS   128
#define OUT_FEATS  16
#define NUM_HEADS  4
#define EDGE_FEATS 32
#define NEG_SLOPE  0.2f
#define HD         (NUM_HEADS * OUT_FEATS)   // 64

typedef __attribute__((ext_vector_type(16))) _Float16 v16h;
typedef __attribute__((ext_vector_type(8)))  float    v8f;

// ---------- monotone float<->uint mapping for atomic max ----------
__device__ __forceinline__ unsigned f2sort(float v) {
  unsigned b = __float_as_uint(v);
  return (b & 0x80000000u) ? ~b : (b | 0x80000000u);
}
__device__ __forceinline__ float sort2f(unsigned u) {
  return __uint_as_float((u & 0x80000000u) ? (u ^ 0x80000000u) : ~u);
}
__device__ __forceinline__ float leaky(float v) {
  return v > 0.0f ? v : NEG_SLOPE * v;
}

// ---------- zero helper (graph-capture-safe init of out/ws) ----------
__global__ void zero_u32_kernel(unsigned* __restrict__ p, long long n) {
  long long i = (long long)blockIdx.x * blockDim.x + threadIdx.x;
  if (i < n) p[i] = 0u;
}

// ---------- ee[t][h] = sum_fe (edge_emb[t] . fc_e_w[h*32+fe]) * attn_e[h][fe]
__global__ void ee_table_kernel(const float* __restrict__ edge_emb,
                                const float* __restrict__ fc_e_w,
                                const float* __restrict__ attn_e,
                                float* __restrict__ eetab) {
  int tid = threadIdx.x;
  if (tid >= 32) return;
  int t = tid >> 2, h = tid & 3;
  const float* em = edge_emb + t * EDGE_FEATS;
  float sum = 0.0f;
  for (int fe = 0; fe < EDGE_FEATS; ++fe) {
    const float* wr = fc_e_w + (size_t)(h * EDGE_FEATS + fe) * EDGE_FEATS;
    float dot = 0.0f;
    #pragma unroll
    for (int c = 0; c < EDGE_FEATS; ++c) dot += em[c] * wr[c];
    sum += dot * attn_e[h * EDGE_FEATS + fe];
  }
  eetab[t * 4 + h] = sum;
}

// ---------- feat = x @ fc_w.T via v_wmma_f32_16x16x32_f16 ----------
// One 16x16 output tile per wave; block = 4 waves = the 4 N-tiles of one M-tile.
__global__ void feat_gemm_wmma(const float* __restrict__ x,
                               const float* __restrict__ fc_w,
                               float* __restrict__ feat, int N) {
  const int wave  = threadIdx.x >> 5;     // 0..3 -> ntile
  const int lane  = threadIdx.x & 31;
  const int mtile = blockIdx.x;
  const int r16   = lane & 15;
  const int khalf = lane >> 4;            // 0 or 1

  int arow   = mtile * 16 + r16;
  int arow_c = arow < N ? arow : 0;       // clamp loads for tail tile
  const float* xrow = x    + (size_t)arow_c * IN_FEATS;
  const float* wrow = fc_w + (size_t)(wave * 16 + r16) * IN_FEATS;

  v8f c = {};
  #pragma unroll
  for (int kt = 0; kt < IN_FEATS / 32; ++kt) {
    // A fragment (16-bit A 16x32 layout): lanes 0-15 K {0..7,16..23}, lanes 16-31 K {8..15,24..31}
    const int ka = kt * 32 + khalf * 8;
    v16h a, b;
    #pragma unroll
    for (int i = 0; i < 8; ++i) {
      a[i]     = (_Float16)xrow[ka + i];
      a[8 + i] = (_Float16)xrow[ka + 16 + i];
    }
    // B fragment (32x16): lanes 0-15 K {0..7,16..23}, lanes 16-31 K {8..15,24..31} (matches A)
    const int kb = kt * 32 + khalf * 8;
    #pragma unroll
    for (int i = 0; i < 8; ++i) {
      b[i]     = (_Float16)wrow[kb + i];
      b[8 + i] = (_Float16)wrow[kb + 16 + i];
    }

    c = __builtin_amdgcn_wmma_f32_16x16x32_f16(false, a, false, b, (short)0, c,
                                               false, false);
  }

  // C/D layout: lanes 0-15 -> M=r, N=lane ; lanes 16-31 -> M=8+r, N=lane-16
  const int col = wave * 16 + r16;
  #pragma unroll
  for (int r = 0; r < 8; ++r) {
    int orow = mtile * 16 + khalf * 8 + r;
    if (orow < N) feat[(size_t)orow * HD + col] = c[r];
  }
}

// ---------- el/er per node ----------
__global__ void node_attn_kernel(const float* __restrict__ feat,
                                 const float* __restrict__ attn_l,
                                 const float* __restrict__ attn_r,
                                 float* __restrict__ el, float* __restrict__ er,
                                 int N) {
  int n = blockIdx.x * blockDim.x + threadIdx.x;
  if (n >= N) return;
  const float* f = feat + (size_t)n * HD;
  #pragma unroll
  for (int h = 0; h < NUM_HEADS; ++h) {
    float sl = 0.0f, sr = 0.0f;
    #pragma unroll
    for (int i = 0; i < OUT_FEATS; ++i) {
      float v = f[h * OUT_FEATS + i];
      sl += v * attn_l[h * OUT_FEATS + i];
      sr += v * attn_r[h * OUT_FEATS + i];
    }
    el[n * 4 + h] = sl;
    er[n * 4 + h] = sr;
  }
}

// ---------- pass 1: segment max of logits ----------
__global__ void edge_max_kernel(const int* __restrict__ src,
                                const int* __restrict__ dst,
                                const int* __restrict__ etype,
                                const float4* __restrict__ el4,
                                const float4* __restrict__ er4,
                                const float4* __restrict__ ee4,
                                unsigned* __restrict__ emax, int E) {
  int e = blockIdx.x * blockDim.x + threadIdx.x;
  if (e >= E) return;
  int s = src[e], d = dst[e], t = etype[e];
  float4 l = el4[s], r = er4[d], ee = ee4[t];
  float v[4] = {leaky(l.x + r.x + ee.x), leaky(l.y + r.y + ee.y),
                leaky(l.z + r.z + ee.z), leaky(l.w + r.w + ee.w)};
  #pragma unroll
  for (int h = 0; h < 4; ++h)
    atomicMax(&emax[(size_t)d * 4 + h], f2sort(v[h]));
}

// ---------- pass 2: denom = segment sum of exp(e - emax[dst]) ----------
__global__ void edge_denom_kernel(const int* __restrict__ src,
                                  const int* __restrict__ dst,
                                  const int* __restrict__ etype,
                                  const float4* __restrict__ el4,
                                  const float4* __restrict__ er4,
                                  const float4* __restrict__ ee4,
                                  const uint4* __restrict__ emax4,
                                  float* __restrict__ denom, int E) {
  int e = blockIdx.x * blockDim.x + threadIdx.x;
  if (e >= E) return;
  int s = src[e], d = dst[e], t = etype[e];
  float4 l = el4[s], r = er4[d], ee = ee4[t];
  uint4 mu = emax4[d];
  float v[4] = {leaky(l.x + r.x + ee.x), leaky(l.y + r.y + ee.y),
                leaky(l.z + r.z + ee.z), leaky(l.w + r.w + ee.w)};
  float m[4] = {sort2f(mu.x), sort2f(mu.y), sort2f(mu.z), sort2f(mu.w)};
  #pragma unroll
  for (int h = 0; h < 4; ++h)
    atomicAdd(&denom[(size_t)d * 4 + h], expf(v[h] - m[h]));
}

// ---------- pass 3: out[dst] += feat[src] * a ; 16 threads per edge ----------
__global__ void edge_scatter_kernel(const int* __restrict__ src,
                                    const int* __restrict__ dst,
                                    const int* __restrict__ etype,
                                    const float* __restrict__ feat,
                                    const float4* __restrict__ el4,
                                    const float4* __restrict__ er4,
                                    const float4* __restrict__ ee4,
                                    const uint4* __restrict__ emax4,
                                    const float4* __restrict__ den4,
                                    float* __restrict__ out, int E) {
  long long gid = (long long)blockIdx.x * blockDim.x + threadIdx.x;
  int e = (int)(gid >> 4);
  int j = (int)(gid & 15);
  if (e >= E) return;
  int s = src[e], d = dst[e], t = etype[e];
  float4 l = el4[s], r = er4[d], ee = ee4[t], den = den4[d];
  uint4 mu = emax4[d];
  float v[4] = {leaky(l.x + r.x + ee.x), leaky(l.y + r.y + ee.y),
                leaky(l.z + r.z + ee.z), leaky(l.w + r.w + ee.w)};
  float m[4]  = {sort2f(mu.x), sort2f(mu.y), sort2f(mu.z), sort2f(mu.w)};
  float dn[4] = {den.x, den.y, den.z, den.w};
  const float* fs = feat + (size_t)s * HD;
  float* od = out + (size_t)d * HD;
  #pragma unroll
  for (int h = 0; h < 4; ++h) {
    float a = expf(v[h] - m[h]) / dn[h];
    atomicAdd(&od[h * OUT_FEATS + j], fs[h * OUT_FEATS + j] * a);
  }
}

extern "C" void kernel_launch(void* const* d_in, const int* in_sizes, int n_in,
                              void* d_out, int out_size, void* d_ws, size_t ws_size,
                              hipStream_t stream) {
  const float* x        = (const float*)d_in[0];
  const int*   src      = (const int*)d_in[1];
  const int*   dst      = (const int*)d_in[2];
  const int*   etype    = (const int*)d_in[3];
  const float* fc_w     = (const float*)d_in[4];
  const float* fc_e_w   = (const float*)d_in[5];
  const float* edge_emb = (const float*)d_in[6];
  const float* attn_l   = (const float*)d_in[7];
  const float* attn_r   = (const float*)d_in[8];
  const float* attn_e   = (const float*)d_in[9];
  float* out = (float*)d_out;

  const int N = in_sizes[0] / IN_FEATS;
  const int E = in_sizes[1];

  // workspace layout (all 16B-aligned given 256B-aligned d_ws)
  float*    feat  = (float*)d_ws;                         // N*64
  float*    el    = feat + (size_t)N * HD;                // N*4
  float*    er    = el   + (size_t)N * 4;                 // N*4
  float*    eetab = er   + (size_t)N * 4;                 // 32
  unsigned* emax  = (unsigned*)(eetab + 32);              // N*4
  float*    denom = (float*)(emax + (size_t)N * 4);       // N*4

  // zero out + (emax|denom are contiguous: 2*N*4 u32)
  {
    long long n_out = (long long)N * HD;
    zero_u32_kernel<<<(unsigned)((n_out + 255) / 256), 256, 0, stream>>>((unsigned*)out, n_out);
    long long n_ws = (long long)N * 8;
    zero_u32_kernel<<<(unsigned)((n_ws + 255) / 256), 256, 0, stream>>>(emax, n_ws);
  }

  ee_table_kernel<<<1, 32, 0, stream>>>(edge_emb, fc_e_w, attn_e, eetab);

  const int mtiles = (N + 15) / 16;
  feat_gemm_wmma<<<mtiles, 128, 0, stream>>>(x, fc_w, feat, N);

  node_attn_kernel<<<(N + 255) / 256, 256, 0, stream>>>(feat, attn_l, attn_r, el, er, N);

  edge_max_kernel<<<(E + 255) / 256, 256, 0, stream>>>(
      src, dst, etype, (const float4*)el, (const float4*)er,
      (const float4*)eetab, emax, E);

  edge_denom_kernel<<<(E + 255) / 256, 256, 0, stream>>>(
      src, dst, etype, (const float4*)el, (const float4*)er,
      (const float4*)eetab, (const uint4*)emax, denom, E);

  long long scat = (long long)E * 16;
  edge_scatter_kernel<<<(unsigned)((scat + 255) / 256), 256, 0, stream>>>(
      src, dst, etype, feat, (const float4*)el, (const float4*)er,
      (const float4*)eetab, (const uint4*)emax, (const float4*)denom, out, E);
}
